// ReweightedUnivariateLoss_55516747268199
// MI455X (gfx1250) — compile-verified, matrix operand order checked by reading, not run on hardware
//
#include <hip/hip_runtime.h>
#include <hip/hip_bf16.h>

typedef __attribute__((ext_vector_type(16))) _Float16 v16h;
typedef __attribute__((ext_vector_type(8)))  float    v8f;

#define NTOT       8192      // n*K = 1024*8
#define ROWS_BLK   128       // 8 waves * 16 rows
#define NBLK       (NTOT / ROWS_BLK)   // 64 blocks covers worst case cntPos = 8192
#define POS_PAD    (1e30f)
#define NEG_PAD    (-1e30f)
#define L2E        1.4426950408889634f   // log2(e)
#define LN2        0.6931471805599453    // ln(2)

// ---------------- workspace layout (bytes) ----------------
// [0,   8)      : uint counts[2]  {cntPos, cntNeg}
// [256, +32KB)  : float posArr[8192]   (padded with +1e30)
// [.. , +32KB)  : float negArr[8192]   (padded with -1e30)
// [.. , +256B)  : float partials[64]   (sums of log2-softplus; scaled by ln2 at finalize)
#define WS_POS_OFF   256
#define WS_NEG_OFF   (WS_POS_OFF + NTOT * 4)
#define WS_PART_OFF  (WS_NEG_OFF + NTOT * 4)

// ---------------------------------------------------------------------------
// Kernel 1: reset workspace (pads + zeroed partials) — deterministic launches.
// ---------------------------------------------------------------------------
__global__ void rul_init_kernel(float* posArr, float* negArr, float* partials,
                                unsigned* counts) {
    int i = blockIdx.x * blockDim.x + threadIdx.x;
    if (i < NTOT) {
        posArr[i] = POS_PAD;
        negArr[i] = NEG_PAD;
    }
    if (i < NBLK) partials[i] = 0.0f;
    if (i < 2)    counts[i]   = 0u;
}

// ---------------------------------------------------------------------------
// Kernel 2: deterministic compaction via LDS prefix scan (1 block, 1024 thr).
// ---------------------------------------------------------------------------
__global__ void rul_compact_kernel(const float* __restrict__ pred,
                                   const int* __restrict__ label,
                                   float* posArr, float* negArr,
                                   unsigned* counts) {
    __shared__ unsigned sPos[1024];
    __shared__ unsigned sNeg[1024];
    const int t    = threadIdx.x;
    const int base = t * 8;

    float lp[8], ln[8];
    unsigned cp = 0, cn = 0;
#pragma unroll
    for (int i = 0; i < 8; ++i) {
        int   y = label[base + i];
        float v = pred[base + i];
        if (y == 1)      lp[cp++] = v;
        else if (y == 0) ln[cn++] = v;
    }
    sPos[t] = cp;
    sNeg[t] = cn;
    __syncthreads();

    // inclusive Hillis-Steele scan over 1024 counts
    for (int off = 1; off < 1024; off <<= 1) {
        unsigned ap = (t >= off) ? sPos[t - off] : 0u;
        unsigned an = (t >= off) ? sNeg[t - off] : 0u;
        __syncthreads();
        sPos[t] += ap;
        sNeg[t] += an;
        __syncthreads();
    }

    unsigned pBase = sPos[t] - cp;
    unsigned nBase = sNeg[t] - cn;
    for (unsigned i = 0; i < cp; ++i) posArr[pBase + i] = lp[i];
    for (unsigned i = 0; i < cn; ++i) negArr[nBase + i] = ln[i];
    if (t == 1023) {
        counts[0] = sPos[1023];
        counts[1] = sNeg[1023];
    }
}

// ---------------------------------------------------------------------------
// Kernel 3: all-pairs softplus through raw base-2 transcendentals, reduced on
// the matrix unit. Per element: v_fma, v_exp_f32, v_add, v_log_f32, then a
// paired v_cvt_pk_f16_f32 — no precise-libm fixup code. The trailing *ln2 is
// linear so it is deferred past the WMMA reduction to the finalize kernel.
// log2-softplus(x) = log2(1 + 2^(vn*L2E - va*L2E)); pads drive it to exact 0.
// ---------------------------------------------------------------------------
__device__ __forceinline__ float log2_softplus(float vn, float vaL2E) {
    float t = __builtin_fmaf(vn, L2E, -vaL2E);     // x * log2(e), one v_fma
    float e = __builtin_amdgcn_exp2f(t);           // v_exp_f32 (2^t)
    return __builtin_amdgcn_logf(1.0f + e);        // v_log_f32 (log2)
}

__global__ void __launch_bounds__(256)
rul_pairs_kernel(const float* __restrict__ posArr,
                 const float* __restrict__ negArr,
                 const unsigned* __restrict__ counts,
                 float* __restrict__ partials) {
    __shared__ float sNeg[NTOT];   // 32 KB of the 320 KB WGP LDS
    __shared__ float sPart[16];

    const unsigned cntPos = counts[0];
    const unsigned cntNeg = counts[1];

    const int rowBlockBase = blockIdx.x * ROWS_BLK;
    if ((unsigned)rowBlockBase >= cntPos) return;   // uniform per block

    const unsigned negElems = ((cntNeg + 31u) >> 5) << 5;  // 32-aligned

    // cooperative stage of the (padded) negative vector into LDS
    for (unsigned i = threadIdx.x; i < negElems; i += 256u)
        sNeg[i] = negArr[i];
    __syncthreads();

    const int warp = threadIdx.x >> 5;
    const int lane = threadIdx.x & 31;
    const bool hi  = lane >= 16;

    // this lane's A-matrix row (pad rows hold +1e30 -> contribution == 0)
    const int   m     = rowBlockBase + warp * 16 + (lane & 15);
    const float vaL2E = posArr[m] * L2E;   // hoisted: folds row term into fma

    v16h ones;
#pragma unroll
    for (int e = 0; e < 16; ++e) ones[e] = (_Float16)1.0f;

    v8f c = {0.f, 0.f, 0.f, 0.f, 0.f, 0.f, 0.f, 0.f};

    const float4* sNeg4 = (const float4*)sNeg;
    const unsigned nChunks = negElems >> 5;
    for (unsigned k0 = 0; k0 < nChunks; ++k0) {
        // lane's K-values: two aligned 8-float runs -> 4x ds_load_b128
        const unsigned c4 = (k0 * 32u + (hi ? 8u : 0u)) >> 2;
        float4 q0 = sNeg4[c4];       // K = base+0 .. base+3
        float4 q1 = sNeg4[c4 + 1];   // K = base+4 .. base+7
        float4 q2 = sNeg4[c4 + 4];   // K = base+16 .. base+19
        float4 q3 = sNeg4[c4 + 5];   // K = base+20 .. base+23

        float vn[16] = {q0.x, q0.y, q0.z, q0.w, q1.x, q1.y, q1.z, q1.w,
                        q2.x, q2.y, q2.z, q2.w, q3.x, q3.y, q3.z, q3.w};
        v16h a;
#pragma unroll
        for (int e = 0; e < 16; ++e)
            a[e] = (_Float16)log2_softplus(vn[e], vaL2E);

        // D = A(16x32 log2-softplus) x B(32x16 ones) + C => replicated row sums
        c = __builtin_amdgcn_wmma_f32_16x16x32_f16(
                false, a, false, ones, (short)0, c, false, false);
    }

    // D layout: lanes 0-15 VGPR r => row r; lanes 16-31 VGPR r => row 8+r.
    // Every column is identical, so column 0 (lanes 0 and 16) suffices.
    float partial = 0.f;
#pragma unroll
    for (int r = 0; r < 8; ++r) partial += c[r];

    if ((lane & 15) == 0) sPart[warp * 2 + (hi ? 1 : 0)] = partial;
    __syncthreads();
    if (threadIdx.x == 0) {
        float t = 0.f;
        for (int i = 0; i < 16; ++i) t += sPart[i];
        partials[blockIdx.x] = t;   // early-exit blocks stay at init's 0
    }
}

// ---------------------------------------------------------------------------
// Kernel 4: serial fixed-order finalize; apply the deferred ln2 scale here.
// ---------------------------------------------------------------------------
__global__ void rul_finalize_kernel(const float* __restrict__ partials,
                                    const unsigned* __restrict__ counts,
                                    float* __restrict__ out) {
    if (threadIdx.x == 0 && blockIdx.x == 0) {
        float total = 0.f;
        for (int i = 0; i < NBLK; ++i) total += partials[i];
        double denom = (double)counts[0] * (double)counts[1];
        out[0] = (float)(LN2 * (double)total / denom);
    }
}

extern "C" void kernel_launch(void* const* d_in, const int* in_sizes, int n_in,
                              void* d_out, int out_size, void* d_ws, size_t ws_size,
                              hipStream_t stream) {
    const float* pred  = (const float*)d_in[0];   // pred_y f32 [8192]
    const int*   label = (const int*)d_in[1];     // true_y i32 [8192]
    float*       out   = (float*)d_out;           // scalar f32

    char* ws = (char*)d_ws;
    unsigned* counts   = (unsigned*)ws;
    float*    posArr   = (float*)(ws + WS_POS_OFF);
    float*    negArr   = (float*)(ws + WS_NEG_OFF);
    float*    partials = (float*)(ws + WS_PART_OFF);

    rul_init_kernel<<<(NTOT + 255) / 256, 256, 0, stream>>>(posArr, negArr,
                                                            partials, counts);
    rul_compact_kernel<<<1, 1024, 0, stream>>>(pred, label, posArr, negArr,
                                               counts);
    rul_pairs_kernel<<<NBLK, 256, 0, stream>>>(posArr, negArr, counts,
                                               partials);
    rul_finalize_kernel<<<1, 32, 0, stream>>>(partials, counts, out);
}
